// Transformer_17282948399785
// MI455X (gfx1250) — matrix operand order repaired and checked by reading, not compiled, not run on hardware
//
#include <hip/hip_runtime.h>
#include <hip/hip_bf16.h>
#include <stdint.h>

#define BB    4
#define TT    1024
#define HH    1024
#define NHH   16
#define LL    8
#define SDD   128
#define ADIM  32
#define CVV   16
#define MAXT  1025
#define DH    64
#define FF    4096
#define MTOK  (BB*MAXT)   /* 4100 valid token rows */
#define MP    4160        /* padded rows, multiple of 64 */
#define TPAD  1056        /* padded keys, multiple of 32 */
#define EPSF  1e-6f

typedef _Float16 f16;
typedef __attribute__((ext_vector_type(16))) _Float16 v16h;
typedef __attribute__((ext_vector_type(8)))  _Float16 v8h;
typedef __attribute__((ext_vector_type(8)))  float    v8f;

#define WMMA_F16(a, b, c) \
  __builtin_amdgcn_wmma_f32_16x16x32_f16(false, (a), false, (b), (short)0, (c), false, false)

// A-fragment: lanes 0-15 row M=lane, K = k..k+15 ; lanes 16-31 same rows,
// K offset by 16 (caller passes ptr already offset by kg16). Two 16B loads.
static __device__ __forceinline__ v16h load_afrag(const f16* p) {
  v8h lo = *(const v8h*)p;
  v8h hi = *(const v8h*)(p + 8);
  v16h r;
#pragma unroll
  for (int i = 0; i < 8; ++i) { r[i] = lo[i]; r[i + 8] = hi[i]; }
  return r;
}

// Async global->LDS 16B copy (per-lane), tracked with ASYNCcnt.
// LDS dest = low 32 bits of the shared-aperture flat address.
static __device__ __forceinline__ void async_ld16(uint32_t ldsoff, const f16* g) {
  asm volatile("global_load_async_to_lds_b128 %0, %1, off"
               :: "v"(ldsoff), "v"((uint64_t)(uintptr_t)g)
               : "memory");
}
static __device__ __forceinline__ void wait_async0() {
  asm volatile("s_wait_asynccnt 0x0" ::: "memory");
}
static __device__ __forceinline__ uint32_t lds_off(const void* p) {
  return (uint32_t)(uintptr_t)p;
}

static __device__ __forceinline__ float gelu_f(float x) {
  float x3 = x * x * x;
  return 0.5f * x * (1.f + tanhf(0.7978845608028654f * (x + 0.044715f * x3)));
}

// ---------------------------------------------------------------------------
// Weight convert + transpose:  W (K x N, fp32 row-major)  ->  Wt (N x K, f16)
// ---------------------------------------------------------------------------
__global__ void convert_wt_kernel(const float* __restrict__ W, f16* __restrict__ Wt,
                                  int K, int N) {
  size_t total = (size_t)K * N;
  for (size_t i = (size_t)blockIdx.x * blockDim.x + threadIdx.x; i < total;
       i += (size_t)gridDim.x * blockDim.x) {
    size_t k = i / N, n = i % N;
    Wt[n * (size_t)K + k] = (f16)W[i];
  }
}

// ---------------------------------------------------------------------------
// Embedding + LN0.  One workgroup (256 thr) per padded row.
// ---------------------------------------------------------------------------
__global__ __launch_bounds__(256) void embed_ln_kernel(
    const float* __restrict__ states, const float* __restrict__ actions,
    const float* __restrict__ Ws, const float* __restrict__ bs,
    const float* __restrict__ Wa, const float* __restrict__ ba,
    const float* __restrict__ pos, const float* __restrict__ g0,
    const float* __restrict__ b0, float* __restrict__ hf32,
    f16* __restrict__ h16) {
  const int row = blockIdx.x;
  const int tid = threadIdx.x;
  __shared__ float red[256];
  if (row >= MTOK) {
#pragma unroll
    for (int j = 0; j < 4; ++j) {
      int c = tid + j * 256;
      hf32[(size_t)row * HH + c] = 0.f;
      h16[(size_t)row * HH + c] = (f16)0.f;
    }
    return;
  }
  const int b = row / MAXT, t = row % MAXT;
  float e[4];
  float lsum = 0.f, lsq = 0.f;
#pragma unroll
  for (int j = 0; j < 4; ++j) {
    const int c = tid + j * 256;
    float acc;
    if (t == 0) {
      acc = bs[c] + pos[c];
      const float* sp = states + (size_t)b * TT * SDD;
      for (int i = 0; i < SDD; ++i) acc += sp[i] * Ws[(size_t)i * HH + c];
    } else {
      acc = ba[c] + pos[(size_t)t * HH + c];
      const float* ap = actions + ((size_t)b * TT + (t - 1)) * ADIM;
      for (int i = 0; i < ADIM; ++i) acc += ap[i] * Wa[(size_t)i * HH + c];
    }
    e[j] = acc;
    lsum += acc;
    lsq += acc * acc;
  }
  red[tid] = lsum; __syncthreads();
  for (int s = 128; s > 0; s >>= 1) { if (tid < s) red[tid] += red[tid + s]; __syncthreads(); }
  float tot = red[0]; __syncthreads();
  red[tid] = lsq; __syncthreads();
  for (int s = 128; s > 0; s >>= 1) { if (tid < s) red[tid] += red[tid + s]; __syncthreads(); }
  float totsq = red[0];
  const float mean = tot / HH;
  const float var = totsq / HH - mean * mean;
  const float inv = rsqrtf(var + EPSF);
#pragma unroll
  for (int j = 0; j < 4; ++j) {
    const int c = tid + j * 256;
    float y = (e[j] - mean) * inv * g0[c] + b0[c];
    hf32[(size_t)row * HH + c] = y;
    h16[(size_t)row * HH + c] = (f16)y;
  }
}

// ---------------------------------------------------------------------------
// Residual add + LN.  x = hf32 + t16 ; LN -> hf32 (new residual) + h16 mirror.
// ---------------------------------------------------------------------------
__global__ __launch_bounds__(256) void add_ln_kernel(
    float* __restrict__ hf32, const f16* __restrict__ t16,
    const float* __restrict__ g, const float* __restrict__ be,
    f16* __restrict__ h16) {
  const int row = blockIdx.x;
  const int tid = threadIdx.x;
  __shared__ float red[256];
  if (row >= MTOK) {
#pragma unroll
    for (int j = 0; j < 4; ++j) {
      int c = tid + j * 256;
      hf32[(size_t)row * HH + c] = 0.f;
      h16[(size_t)row * HH + c] = (f16)0.f;
    }
    return;
  }
  float x[4];
  float lsum = 0.f, lsq = 0.f;
#pragma unroll
  for (int j = 0; j < 4; ++j) {
    const int c = tid + j * 256;
    float v = hf32[(size_t)row * HH + c] + (float)t16[(size_t)row * HH + c];
    x[j] = v;
    lsum += v;
    lsq += v * v;
  }
  red[tid] = lsum; __syncthreads();
  for (int s = 128; s > 0; s >>= 1) { if (tid < s) red[tid] += red[tid + s]; __syncthreads(); }
  float tot = red[0]; __syncthreads();
  red[tid] = lsq; __syncthreads();
  for (int s = 128; s > 0; s >>= 1) { if (tid < s) red[tid] += red[tid + s]; __syncthreads(); }
  float totsq = red[0];
  const float mean = tot / HH;
  const float var = totsq / HH - mean * mean;
  const float inv = rsqrtf(var + EPSF);
#pragma unroll
  for (int j = 0; j < 4; ++j) {
    const int c = tid + j * 256;
    float y = (x[j] - mean) * inv * g[c] + be[c];
    hf32[(size_t)row * HH + c] = y;
    h16[(size_t)row * HH + c] = (f16)y;
  }
}

// ---------------------------------------------------------------------------
// WMMA GEMM:  C[MP x N] = A[MP x K] * Wt[N x K]^T + bias, optional GELU, f16 out.
// 128 threads = 4 waves; block tile 64x64, wave w owns a 64x16 strip.
// A tile (64x32) staged once per block into LDS via async global->LDS loads,
// double-buffered so the next tile's copies overlap the current tile's WMMAs.
// ---------------------------------------------------------------------------
__global__ __launch_bounds__(128) void gemm_bias_kernel(
    const f16* __restrict__ A, const f16* __restrict__ Wt,
    const float* __restrict__ bias, f16* __restrict__ C,
    int N, int K, int do_gelu) {
  __shared__ f16 asmem[2][64 * 32];   // 2 x 4KB double buffer
  const int lane = threadIdx.x & 31;
  const int wave = threadIdx.x >> 5;
  const int l16 = lane & 15;
  const int hi8 = (lane >> 4) << 3;    // C row offset
  const int kg16 = (lane >> 4) << 4;   // B-frag K offset
  const int n0 = blockIdx.x * 64 + wave * 16;
  const int m0 = blockIdx.y * 64;

  const f16* wrow = Wt + (size_t)(n0 + l16) * K;

  // Async staging: 64 rows x 32 halves = 256 x 16B chunks; 2 chunks per thread.
  const int c0i = threadIdx.x;
  const int c1i = threadIdx.x + 128;
  const int row0 = c0i >> 2, kl0 = (c0i & 3) << 3;
  const int row1 = c1i >> 2, kl1 = (c1i & 3) << 3;
  const f16* ga0 = A + (size_t)(m0 + row0) * K + kl0;
  const f16* ga1 = A + (size_t)(m0 + row1) * K + kl1;
  const uint32_t ld0[2] = { lds_off(&asmem[0][row0 * 32 + kl0]),
                            lds_off(&asmem[1][row0 * 32 + kl0]) };
  const uint32_t ld1[2] = { lds_off(&asmem[0][row1 * 32 + kl1]),
                            lds_off(&asmem[1][row1 * 32 + kl1]) };

  // Prime buffer 0.
  async_ld16(ld0[0], ga0);
  async_ld16(ld1[0], ga1);
  wait_async0();
  __syncthreads();

  v8f c0 = {}, c1 = {}, c2 = {}, c3 = {};
  int buf = 0;
  for (int k = 0; k < K; k += 32) {
    if (k + 32 < K) {               // prefetch next A tile into the other buffer
      async_ld16(ld0[buf ^ 1], ga0 + k + 32);
      async_ld16(ld1[buf ^ 1], ga1 + k + 32);
    }
    v16h bf = *(const v16h*)(wrow + k + kg16);
    const f16* as = asmem[buf];
    v16h f0 = load_afrag(as + (0 + l16) * 32 + kg16);
    v16h f1 = load_afrag(as + (16 + l16) * 32 + kg16);
    v16h f2 = load_afrag(as + (32 + l16) * 32 + kg16);
    v16h f3 = load_afrag(as + (48 + l16) * 32 + kg16);
    c0 = WMMA_F16(f0, bf, c0);
    c1 = WMMA_F16(f1, bf, c1);
    c2 = WMMA_F16(f2, bf, c2);
    c3 = WMMA_F16(f3, bf, c3);
    wait_async0();                  // next buffer landed
    __syncthreads();                // everyone done reading current buffer
    buf ^= 1;
  }

  const int col = n0 + l16;
  const float bval = bias[col];
#pragma unroll
  for (int r = 0; r < 8; ++r) {
    float v0 = c0[r] + bval, v1 = c1[r] + bval, v2 = c2[r] + bval, v3 = c3[r] + bval;
    if (do_gelu) { v0 = gelu_f(v0); v1 = gelu_f(v1); v2 = gelu_f(v2); v3 = gelu_f(v3); }
    const int rr = hi8 + r;
    C[(size_t)(m0 + 0 + rr) * N + col] = (f16)v0;
    C[(size_t)(m0 + 16 + rr) * N + col] = (f16)v1;
    C[(size_t)(m0 + 32 + rr) * N + col] = (f16)v2;
    C[(size_t)(m0 + 48 + rr) * N + col] = (f16)v3;
  }
}

// ---------------------------------------------------------------------------
// V transpose for attention: vt[b][h][d][t] = V[b*MAXT+t][h*64+d], zero padded.
// ---------------------------------------------------------------------------
__global__ void transpose_v_kernel(const f16* __restrict__ vbuf, f16* __restrict__ vt) {
  size_t total = (size_t)BB * NHH * DH * TPAD;
  for (size_t i = (size_t)blockIdx.x * blockDim.x + threadIdx.x; i < total;
       i += (size_t)gridDim.x * blockDim.x) {
    int t = (int)(i % TPAD);
    size_t rem = i / TPAD;
    int d = (int)(rem % DH); rem /= DH;
    int h = (int)(rem % NHH);
    int b = (int)(rem / NHH);
    f16 val = (f16)0.f;
    if (t < MAXT) val = vbuf[((size_t)(b * MAXT + t)) * HH + h * DH + d];
    vt[i] = val;
  }
}

// ---------------------------------------------------------------------------
// Flash attention: grid (B*NH, 17), 4 waves/block, wave = one 16-query tile.
// ---------------------------------------------------------------------------
__global__ __launch_bounds__(128) void attention_kernel(
    const f16* __restrict__ q16, const f16* __restrict__ k16,
    const f16* __restrict__ vt, f16* __restrict__ o16) {
  const int lane = threadIdx.x & 31;
  const int wave = threadIdx.x >> 5;
  const int l16 = lane & 15;
  const int hi8 = (lane >> 4) << 3;
  const int kg16 = (lane >> 4) << 4;
  const int bh = blockIdx.x;
  const int b = bh / NHH, h = bh % NHH;
  const int qt = blockIdx.y * 4 + wave;
  __shared__ f16 pstage[4][16 * 32];
  if (qt > 64) return;
  const int qbase = qt * 16;

  int tq = qbase + l16; if (tq > 1024) tq = 1024;
  const f16* qrow = q16 + ((size_t)(b * MAXT + tq)) * HH + h * DH;
  const v16h aq0 = load_afrag(qrow + kg16);
  const v16h aq1 = load_afrag(qrow + 32 + kg16);

  float mrow[8], lrow[8];
  v8f o0 = {}, o1 = {}, o2 = {}, o3 = {};
#pragma unroll
  for (int r = 0; r < 8; ++r) { mrow[r] = -1e30f; lrow[r] = 0.f; }

  int qmax = qbase + 15; if (qmax > 1024) qmax = 1024;
  const int nkt = qmax / 32 + 1;
  f16* pst = pstage[wave];

  for (int kt = 0; kt < nkt; ++kt) {
    const int jb = kt * 32;
    const int j0 = jb + l16;
    const int j1 = jb + 16 + l16;
    const int j0c = j0 > 1024 ? 1024 : j0;
    const int j1c = j1 > 1024 ? 1024 : j1;
    const f16* kr0 = k16 + ((size_t)(b * MAXT + j0c)) * HH + h * DH;
    const f16* kr1 = k16 + ((size_t)(b * MAXT + j1c)) * HH + h * DH;
    v16h bk00 = *(const v16h*)(kr0 + kg16);
    v16h bk01 = *(const v16h*)(kr0 + 32 + kg16);
    v16h bk10 = *(const v16h*)(kr1 + kg16);
    v16h bk11 = *(const v16h*)(kr1 + 32 + kg16);
    v8f s0 = {}, s1 = {};
    s0 = WMMA_F16(aq0, bk00, s0);
    s0 = WMMA_F16(aq1, bk01, s0);
    s1 = WMMA_F16(aq0, bk10, s1);
    s1 = WMMA_F16(aq1, bk11, s1);

#pragma unroll
    for (int r = 0; r < 8; ++r) {
      const int trow = qbase + r + hi8;
      float sv0 = (j0 <= trow) ? s0[r] * 0.125f : -1e30f;
      float sv1 = (j1 <= trow) ? s1[r] * 0.125f : -1e30f;
      float mx = fmaxf(sv0, sv1);
      mx = fmaxf(mx, __shfl_xor(mx, 1, 32));
      mx = fmaxf(mx, __shfl_xor(mx, 2, 32));
      mx = fmaxf(mx, __shfl_xor(mx, 4, 32));
      mx = fmaxf(mx, __shfl_xor(mx, 8, 32));
      const float nm = fmaxf(mrow[r], mx);
      const float alpha = __expf(mrow[r] - nm);
      const float p0 = __expf(sv0 - nm);
      const float p1 = __expf(sv1 - nm);
      float rs = p0 + p1;
      rs += __shfl_xor(rs, 1, 32);
      rs += __shfl_xor(rs, 2, 32);
      rs += __shfl_xor(rs, 4, 32);
      rs += __shfl_xor(rs, 8, 32);
      lrow[r] = lrow[r] * alpha + rs;
      mrow[r] = nm;
      o0[r] *= alpha; o1[r] *= alpha; o2[r] *= alpha; o3[r] *= alpha;
      pst[(r + hi8) * 32 + l16] = (f16)p0;
      pst[(r + hi8) * 32 + 16 + l16] = (f16)p1;
    }
    asm volatile("s_wait_dscnt 0x0" ::: "memory");
    const v16h ap = load_afrag(pst + l16 * 32 + kg16);

    const f16* vr0 = vt + ((size_t)(bh * DH + 0 + l16)) * TPAD + jb + kg16;
    const f16* vr1 = vt + ((size_t)(bh * DH + 16 + l16)) * TPAD + jb + kg16;
    const f16* vr2 = vt + ((size_t)(bh * DH + 32 + l16)) * TPAD + jb + kg16;
    const f16* vr3 = vt + ((size_t)(bh * DH + 48 + l16)) * TPAD + jb + kg16;
    o0 = WMMA_F16(ap, *(const v16h*)vr0, o0);
    o1 = WMMA_F16(ap, *(const v16h*)vr1, o1);
    o2 = WMMA_F16(ap, *(const v16h*)vr2, o2);
    o3 = WMMA_F16(ap, *(const v16h*)vr3, o3);
  }

#pragma unroll
  for (int r = 0; r < 8; ++r) {
    const int trow = qbase + r + hi8;
    if (trow <= 1024) {
      const float invl = 1.f / lrow[r];
      const size_t base = ((size_t)(b * MAXT + trow)) * HH + h * DH;
      o16[base + 0 + l16] = (f16)(o0[r] * invl);
      o16[base + 16 + l16] = (f16)(o1[r] * invl);
      o16[base + 32 + l16] = (f16)(o2[r] * invl);
      o16[base + 48 + l16] = (f16)(o3[r] * invl);
    }
  }
}

// ---------------------------------------------------------------------------
// Output head: out[b,t,:] = h[b, t+1, :] @ W_out + b_out   (fp32, 32 cols)
// ---------------------------------------------------------------------------
__global__ __launch_bounds__(128) void head_kernel(
    const float* __restrict__ hf32, const float* __restrict__ Wout,
    const float* __restrict__ bout, float* __restrict__ out) {
  const int lane = threadIdx.x & 31;
  const int wave = threadIdx.x >> 5;
  const int rowid = blockIdx.x * 4 + wave;
  if (rowid >= BB * TT) return;
  const int b = rowid >> 10, t = rowid & 1023;
  const float* hr = hf32 + ((size_t)(b * MAXT + t + 1)) * HH;
  float acc = bout[lane];
  for (int k = 0; k < HH; ++k) acc += hr[k] * Wout[(size_t)k * (2 * CVV) + lane];
  out[(size_t)rowid * (2 * CVV) + lane] = acc;
}

// ---------------------------------------------------------------------------
extern "C" void kernel_launch(void* const* d_in, const int* in_sizes, int n_in,
                              void* d_out, int out_size, void* d_ws, size_t ws_size,
                              hipStream_t stream) {
  const float* states = (const float*)d_in[1];
  const float* actions = (const float*)d_in[3];
  const float* W_s = (const float*)d_in[7];
  const float* b_s = (const float*)d_in[8];
  const float* W_a = (const float*)d_in[9];
  const float* b_a = (const float*)d_in[10];
  const float* pos = (const float*)d_in[11];
  const float* ln0_s = (const float*)d_in[12];
  const float* ln0_b = (const float*)d_in[13];
  const float* Wq = (const float*)d_in[14];
  const float* bq = (const float*)d_in[15];
  const float* Wk = (const float*)d_in[16];
  const float* bk = (const float*)d_in[17];
  const float* Wv = (const float*)d_in[18];
  const float* bv = (const float*)d_in[19];
  const float* Wp = (const float*)d_in[20];
  const float* bp = (const float*)d_in[21];
  const float* ln1_s = (const float*)d_in[22];
  const float* ln1_b = (const float*)d_in[23];
  const float* W1 = (const float*)d_in[24];
  const float* b1 = (const float*)d_in[25];
  const float* W2 = (const float*)d_in[26];
  const float* b2 = (const float*)d_in[27];
  const float* ln2_s = (const float*)d_in[28];
  const float* ln2_b = (const float*)d_in[29];
  const float* W_out = (const float*)d_in[30];
  const float* b_out = (const float*)d_in[31];

  char* wp = (char*)d_ws;
  auto alloc = [&](size_t bytes) -> char* {
    char* r = wp;
    wp += (bytes + 255) & ~(size_t)255;
    return r;
  };
  float* hf32 = (float*)alloc((size_t)MP * HH * 4);
  f16* h16 = (f16*)alloc((size_t)MP * HH * 2);
  f16* t16 = (f16*)alloc((size_t)MP * HH * 2);
  f16* q16 = (f16*)alloc((size_t)MP * HH * 2);
  f16* k16 = (f16*)alloc((size_t)MP * HH * 2);
  f16* v16b = (f16*)alloc((size_t)MP * HH * 2);
  f16* o16 = (f16*)alloc((size_t)MP * HH * 2);
  f16* g16 = (f16*)alloc((size_t)MP * FF * 2);
  f16* vt = (f16*)alloc((size_t)BB * NHH * DH * TPAD * 2);
  f16* wq16 = (f16*)alloc((size_t)LL * HH * HH * 2);
  f16* wk16 = (f16*)alloc((size_t)LL * HH * HH * 2);
  f16* wv16 = (f16*)alloc((size_t)LL * HH * HH * 2);
  f16* wp16 = (f16*)alloc((size_t)LL * HH * HH * 2);
  f16* w116 = (f16*)alloc((size_t)LL * HH * FF * 2);
  f16* w216 = (f16*)alloc((size_t)LL * FF * HH * 2);

  // Weight conversion (f32 -> f16, transposed to N x K).
  for (int l = 0; l < LL; ++l) {
    size_t oHH = (size_t)l * HH * HH;
    size_t oHF = (size_t)l * HH * FF;
    convert_wt_kernel<<<4096, 256, 0, stream>>>(Wq + oHH, wq16 + oHH, HH, HH);
    convert_wt_kernel<<<4096, 256, 0, stream>>>(Wk + oHH, wk16 + oHH, HH, HH);
    convert_wt_kernel<<<4096, 256, 0, stream>>>(Wv + oHH, wv16 + oHH, HH, HH);
    convert_wt_kernel<<<4096, 256, 0, stream>>>(Wp + oHH, wp16 + oHH, HH, HH);
    convert_wt_kernel<<<8192, 256, 0, stream>>>(W1 + oHF, w116 + oHF, HH, FF);
    convert_wt_kernel<<<8192, 256, 0, stream>>>(W2 + oHF, w216 + oHF, FF, HH);
  }

  embed_ln_kernel<<<MP, 256, 0, stream>>>(states, actions, W_s, b_s, W_a, b_a,
                                          pos, ln0_s, ln0_b, hf32, h16);

  const dim3 gH(HH / 64, MP / 64);
  const dim3 gF(FF / 64, MP / 64);
  for (int l = 0; l < LL; ++l) {
    size_t oHH = (size_t)l * HH * HH;
    size_t oH = (size_t)l * HH;
    size_t oF = (size_t)l * FF;
    size_t oHF = (size_t)l * HH * FF;
    gemm_bias_kernel<<<gH, 128, 0, stream>>>(h16, wq16 + oHH, bq + oH, q16, HH, HH, 0);
    gemm_bias_kernel<<<gH, 128, 0, stream>>>(h16, wk16 + oHH, bk + oH, k16, HH, HH, 0);
    gemm_bias_kernel<<<gH, 128, 0, stream>>>(h16, wv16 + oHH, bv + oH, v16b, HH, HH, 0);
    transpose_v_kernel<<<4096, 256, 0, stream>>>(v16b, vt);
    attention_kernel<<<dim3(BB * NHH, 17), 128, 0, stream>>>(q16, k16, vt, o16);
    gemm_bias_kernel<<<gH, 128, 0, stream>>>(o16, wp16 + oHH, bp + oH, t16, HH, HH, 0);
    add_ln_kernel<<<MP, 256, 0, stream>>>(hf32, t16, ln1_s + oH, ln1_b + oH, h16);
    gemm_bias_kernel<<<gF, 128, 0, stream>>>(h16, w116 + oHF, b1 + oF, g16, FF, HH, 1);
    gemm_bias_kernel<<<gH, 128, 0, stream>>>(g16, w216 + oHF, b2 + oH, t16, HH, FF, 0);
    add_ln_kernel<<<MP, 256, 0, stream>>>(hf32, t16, ln2_s + oH, ln2_b + oH, h16);
  }

  head_kernel<<<(BB * TT) / 4, 128, 0, stream>>>(hf32, W_out, b_out, (float*)d_out);
}
